// GAT_20023137534836
// MI455X (gfx1250) — matrix operand order repaired and checked by reading, not compiled, not run on hardware
//
#include <hip/hip_runtime.h>
#include <hip/hip_bf16.h>

// ---------------------------------------------------------------------------
// GAT layer for MI455X (gfx1250, wave32, WMMA).
// B=8, N=2048, F_in=F_out=128.
// Dominant cost: one pass over adj (134 MB) fused with online softmax and
// v_wmma_f32_16x16x32_f16 accumulation (flash-attention style).
// - Softmax in log2 domain (scores prescaled by log2(e)) -> raw v_exp_f32.
// - Accumulator rescale is EXEC-masked; only runs when the row max grows.
// - adj is loaded with TH=NT (streamed once; keeps whT/wh2 resident in the
//   192 MB L2), output stored with TH=NT (written once, never re-read).
// ---------------------------------------------------------------------------

typedef __attribute__((ext_vector_type(16))) _Float16 v16h;
typedef __attribute__((ext_vector_type(8)))  float    v8f;
typedef __attribute__((ext_vector_type(4)))  float    v4f;
typedef __attribute__((ext_vector_type(4)))  int      v4i;

#define B_    8
#define N_    2048
#define F_    128
#define ALPHA 0.2f
#define NEGV  -9e15f
#define LOG2E 1.4426950408889634f

__device__ __forceinline__ float leaky_f(float x) {
  return x >= 0.f ? x : ALPHA * x;
}

// cross-lane gather within the wave32 (dst[lane] = src[srcLane])
__device__ __forceinline__ float bperm_f(int srcLane, float x) {
  return __int_as_float(
      __builtin_amdgcn_ds_bpermute(srcLane << 2, __float_as_int(x)));
}

// ---------------------------------------------------------------------------
// Kernel 0: u1 = W^T a1, u2 = W^T a2, c1 = b.a1, c2 = b.a2, and W16 = f16(W).
// (per-row scores Wh1[n] = h[n].u1 + c1 need no materialized Wh)
// ---------------------------------------------------------------------------
__global__ void __launch_bounds__(128)
gat_prep_kernel(const float* __restrict__ Ww, const float* __restrict__ Wb,
                const float* __restrict__ aw,
                float* __restrict__ u1, float* __restrict__ u2,
                float* __restrict__ c, _Float16* __restrict__ W16) {
  const int t = threadIdx.x;  // 0..127
  // column t of W dotted with both attention halves
  float s1 = 0.f, s2 = 0.f;
  for (int o = 0; o < F_; ++o) {
    const float w = Ww[o * F_ + t];
    s1 += w * aw[o];
    s2 += w * aw[F_ + o];
  }
  u1[t] = s1;
  u2[t] = s2;
  // row t of W converted to f16 once (so GEMM waves never re-convert W)
  for (int f = 0; f < F_; f += 4) {
    const v4f w = *(const v4f*)(Ww + t * F_ + f);
    union { _Float16 e[4]; float2 f2; } pk;
    pk.e[0] = (_Float16)w.x; pk.e[1] = (_Float16)w.y;
    pk.e[2] = (_Float16)w.z; pk.e[3] = (_Float16)w.w;
    *(float2*)(W16 + t * F_ + f) = pk.f2;
  }
  if (t == 0) {
    float c1 = 0.f, c2 = 0.f;
    for (int o = 0; o < F_; ++o) {
      c1 += Wb[o] * aw[o];
      c2 += Wb[o] * aw[F_ + o];
    }
    c[0] = c1;
    c[1] = c2;
  }
}

// ---------------------------------------------------------------------------
// Kernel 1: Wh = h @ W^T + b   via WMMA f16 (f32 accumulate).
// One wave = one 16-row n-tile, all 128 output features (8 C tiles).
// Epilogue writes whT (f16, [B,128,N]) so the attention kernel's B-matrix
// tiles are contiguous per lane.
// A layout (16-bit A 16x32): lane(<16) row=lane, K = {0..7, 16..23};
//                            lane(>=16) row=lane-16, K = {8..15, 24..31}.
// B layout (16-bit B 32x16): lanes 0-15 col=lane, K={0..7,16..23}; lanes 16-31 K={8..15,24..31}.
// ---------------------------------------------------------------------------
__global__ void __launch_bounds__(128)
gat_wh_kernel(const float* __restrict__ h, const _Float16* __restrict__ W16,
              const float* __restrict__ Wb, _Float16* __restrict__ whT) {
  const int lane = threadIdx.x & 31;
  const int wave = threadIdx.x >> 5;
  const int tile = blockIdx.x * 4 + wave;     // 1024 tiles total
  const int b    = tile >> 7;                 // N/16 = 128 tiles per batch
  const int n0   = (tile & 127) << 4;
  const int row  = lane & 15;
  const int hi   = lane >> 4;

  const float* hrow = h + ((size_t)(b * N_) + n0 + row) * F_;

  v8f acc[8];
  const v8f zero = {0.f, 0.f, 0.f, 0.f, 0.f, 0.f, 0.f, 0.f};
#pragma unroll
  for (int t = 0; t < 8; ++t) acc[t] = zero;

#pragma unroll
  for (int kc = 0; kc < 4; ++kc) {
    const int kb = kc * 32;
    v16h A;
    {
      const v4f x0 = *(const v4f*)(hrow + kb + hi * 8);
      const v4f x1 = *(const v4f*)(hrow + kb + hi * 8 + 4);
      const v4f y0 = *(const v4f*)(hrow + kb + 16 + hi * 8);
      const v4f y1 = *(const v4f*)(hrow + kb + 16 + hi * 8 + 4);
      A[0] = (_Float16)x0.x;  A[1] = (_Float16)x0.y;
      A[2] = (_Float16)x0.z;  A[3] = (_Float16)x0.w;
      A[4] = (_Float16)x1.x;  A[5] = (_Float16)x1.y;
      A[6] = (_Float16)x1.z;  A[7] = (_Float16)x1.w;
      A[8] = (_Float16)y0.x;  A[9] = (_Float16)y0.y;
      A[10] = (_Float16)y0.z; A[11] = (_Float16)y0.w;
      A[12] = (_Float16)y1.x; A[13] = (_Float16)y1.y;
      A[14] = (_Float16)y1.z; A[15] = (_Float16)y1.w;
    }
#pragma unroll
    for (int ot = 0; ot < 8; ++ot) {
      // B[k, o] = W[o, k]; lane col o = ot*16+row, K = kb + {8*hi+0..7, 16+8*hi+0..7}
      const _Float16* wrow = W16 + (ot * 16 + row) * F_ + kb + hi * 8;
      union { v16h v; v4f f[2]; } bu;
      bu.f[0] = *(const v4f*)(wrow);
      bu.f[1] = *(const v4f*)(wrow + 16);
      acc[ot] = __builtin_amdgcn_wmma_f32_16x16x32_f16(
          false, A, false, bu.v, (short)0, acc[ot], false, false);
    }
  }

  // D layout: VGPR v holds row M = v + 8*hi, col o = ot*16 + (lane&15).
  // whT[b, o, n]: lane writes 8 consecutive n (one 16-byte store per tile).
#pragma unroll
  for (int ot = 0; ot < 8; ++ot) {
    const int o = ot * 16 + row;
    const float bias = Wb[o];
    union { _Float16 e[8]; v4f f4; } pk;
#pragma unroll
    for (int v = 0; v < 8; ++v) pk.e[v] = (_Float16)(acc[ot][v] + bias);
    *(v4f*)(whT + ((size_t)(b * F_) + o) * N_ + n0 + hi * 8) = pk.f4;
  }
}

// ---------------------------------------------------------------------------
// Kernel 2: per-row additive score halves, prescaled to log2 domain:
//   wh1[n] = (h[n].u1 + c1) * log2(e)   (same for wh2)
// Valid because leaky(c*x) = c*leaky(x) for c > 0, so
//   exp(leaky(s)) = exp2(leaky(s * log2e)).
// ---------------------------------------------------------------------------
__global__ void __launch_bounds__(256)
gat_score_kernel(const float* __restrict__ h, const float* __restrict__ u1,
                 const float* __restrict__ u2, const float* __restrict__ c,
                 float* __restrict__ wh1, float* __restrict__ wh2) {
  const int n = blockIdx.x * 256 + threadIdx.x;   // 0 .. B*N-1
  const v4f* hp = (const v4f*)(h + (size_t)n * F_);
  const v4f* p1 = (const v4f*)u1;
  const v4f* p2 = (const v4f*)u2;
  float s1 = 0.f, s2 = 0.f;
#pragma unroll 8
  for (int q = 0; q < F_ / 4; ++q) {
    const v4f x = hp[q];
    const v4f a = p1[q];
    const v4f d = p2[q];
    s1 += x.x * a.x + x.y * a.y + x.z * a.z + x.w * a.w;
    s2 += x.x * d.x + x.y * d.y + x.z * d.z + x.w * d.w;
  }
  wh1[n] = (s1 + c[0]) * LOG2E;
  wh2[n] = (s2 + c[1]) * LOG2E;
}

// ---------------------------------------------------------------------------
// Kernel 3: fused masked-softmax(attention) @ Wh + LayerNorm + leaky.
// One wave per 16-row i-tile; single pass over adj with online softmax.
// Per 32-j chunk: 8x v_wmma_f32_16x16x32_f16 (P tile x Wh tile).
// adj loads are non-temporal: 134 MB streamed once must not evict the hot
// whT/wh2 working set from the 192 MB L2.
// ---------------------------------------------------------------------------
__global__ void __launch_bounds__(128)
gat_attn_kernel(const int* __restrict__ adj, const float* __restrict__ wh1,
                const float* __restrict__ wh2, const _Float16* __restrict__ whT,
                float* __restrict__ out) {
  __shared__ __align__(16) float smem[4][16][F_];   // 32 KB: per-wave h' tiles

  const int lane = threadIdx.x & 31;
  const int wave = threadIdx.x >> 5;
  const int tile = blockIdx.x * 4 + wave;
  const int b    = tile >> 7;
  const int i0   = (tile & 127) << 4;
  const int row  = lane & 15;   // this lane's attention row (A-matrix layout)
  const int hi   = lane >> 4;

  const float     wh1_i  = wh1[b * N_ + i0 + row];
  const int*      adjrow = adj + ((size_t)(b * N_) + i0 + row) * N_;
  const float*    wh2b   = wh2 + b * N_;
  const _Float16* whTb   = whT + (size_t)b * F_ * N_;

  float m = -3.0e38f;   // running row max (log2 domain)
  float l = 0.f;        // running row expsum (this lane's j-subset)
  v8f acc[8];
  const v8f zero = {0.f, 0.f, 0.f, 0.f, 0.f, 0.f, 0.f, 0.f};
#pragma unroll
  for (int t = 0; t < 8; ++t) acc[t] = zero;

  for (int j0 = 0; j0 < N_; j0 += 32) {
    // A-layout K order for this lane: {8*hi+0..7} then {16+8*hi+0..7}
    const int ja = j0 + hi * 8;
    const int jb = ja + 16;
    // TH=NT: streamed exactly once, keep out of the L2 working set
    const v4i a0 = __builtin_nontemporal_load((const v4i*)(adjrow + ja));
    const v4i a1 = __builtin_nontemporal_load((const v4i*)(adjrow + ja + 4));
    const v4i a2 = __builtin_nontemporal_load((const v4i*)(adjrow + jb));
    const v4i a3 = __builtin_nontemporal_load((const v4i*)(adjrow + jb + 4));
    const v4f w0 = *(const v4f*)(wh2b + ja);
    const v4f w1 = *(const v4f*)(wh2b + ja + 4);
    const v4f w2 = *(const v4f*)(wh2b + jb);
    const v4f w3 = *(const v4f*)(wh2b + jb + 4);
    if (j0 + 32 < N_) __builtin_prefetch(adjrow + ja + 32, 0, 1);

    float s[16];
#define GAT_SCORE(idx, wv, av)                         \
    {                                                  \
      const float sc = leaky_f(wh1_i + (wv));          \
      s[idx] = ((av) == 0) ? NEGV : sc;                \
    }
    GAT_SCORE(0,  w0.x, a0.x) GAT_SCORE(1,  w0.y, a0.y)
    GAT_SCORE(2,  w0.z, a0.z) GAT_SCORE(3,  w0.w, a0.w)
    GAT_SCORE(4,  w1.x, a1.x) GAT_SCORE(5,  w1.y, a1.y)
    GAT_SCORE(6,  w1.z, a1.z) GAT_SCORE(7,  w1.w, a1.w)
    GAT_SCORE(8,  w2.x, a2.x) GAT_SCORE(9,  w2.y, a2.y)
    GAT_SCORE(10, w2.z, a2.z) GAT_SCORE(11, w2.w, a2.w)
    GAT_SCORE(12, w3.x, a3.x) GAT_SCORE(13, w3.y, a3.y)
    GAT_SCORE(14, w3.z, a3.z) GAT_SCORE(15, w3.w, a3.w)
#undef GAT_SCORE

    float mx = s[0];
#pragma unroll
    for (int e = 1; e < 16; ++e) mx = fmaxf(mx, s[e]);
    // lanes (r, r+16) share row r: combine halves
    mx = fmaxf(mx, bperm_f(lane ^ 16, mx));
    const float mnew = fmaxf(m, mx);

    // EXEC-masked rescale: only the rare chunks that raise the row max pay
    // for the 64-register accumulator rescale (EXEC restored before WMMA).
    if (mnew > m) {
      const float rscale = __builtin_amdgcn_exp2f(m - mnew);
      l *= rscale;
#pragma unroll
      for (int t = 0; t < 8; ++t) acc[t] *= rscale;
      m = mnew;
    }

    float psum = 0.f;
    v16h P;
#pragma unroll
    for (int e = 0; e < 16; ++e) {
      const float p = __builtin_amdgcn_exp2f(s[e] - m);   // raw v_exp_f32
      psum += p;
      P[e] = (_Float16)p;
    }
    l += psum;

#pragma unroll
    for (int ot = 0; ot < 8; ++ot) {
      // B tile: col o = ot*16+row, K = j0 + {8*hi+0..7, 16+8*hi+0..7}
      const _Float16* bp =
          whTb + (size_t)(ot * 16 + row) * N_ + j0 + hi * 8;
      union { v16h v; v4f f[2]; } bu;
      bu.f[0] = *(const v4f*)(bp);
      bu.f[1] = *(const v4f*)(bp + 16);
      acc[ot] = __builtin_amdgcn_wmma_f32_16x16x32_f16(
          false, P, false, bu.v, (short)0, acc[ot], false, false);
    }
  }

  // total row expsum and 1/l; broadcast per accumulator row M = v + 8*hi
  l += bperm_f(lane ^ 16, l);
  const float inv = 1.0f / l;
#pragma unroll
  for (int v = 0; v < 8; ++v) {
    const float invv = bperm_f(v + (hi << 3), inv);
#pragma unroll
    for (int t = 0; t < 8; ++t) acc[t][v] *= invv;
  }

  // spill h' tile to LDS for row-major LayerNorm
#pragma unroll
  for (int t = 0; t < 8; ++t)
#pragma unroll
    for (int v = 0; v < 8; ++v)
      smem[wave][v + (hi << 3)][t * 16 + row] = acc[t][v];

  __syncthreads();

  // LayerNorm + leaky: lane handles row `row`, columns [hi*64, hi*64+64)
  const float* rp = &smem[wave][row][hi * 64];
  v4f xs[16];
  float sum = 0.f, sq = 0.f;
#pragma unroll
  for (int q = 0; q < 16; ++q) {
    const v4f x = ((const v4f*)rp)[q];
    xs[q] = x;
    sum += x.x + x.y + x.z + x.w;
    sq  += x.x * x.x + x.y * x.y + x.z * x.z + x.w * x.w;
  }
  sum += bperm_f(lane ^ 16, sum);
  sq  += bperm_f(lane ^ 16, sq);
  const float mean = sum * (1.0f / F_);
  const float var  = sq * (1.0f / F_) - mean * mean;
  const float isd  = rsqrtf(var + 1e-5f);

  float* op = out + ((size_t)(b * N_) + i0 + row) * F_ + hi * 64;
#pragma unroll
  for (int q = 0; q < 16; ++q) {
    v4f y;
    y.x = leaky_f((xs[q].x - mean) * isd);
    y.y = leaky_f((xs[q].y - mean) * isd);
    y.z = leaky_f((xs[q].z - mean) * isd);
    y.w = leaky_f((xs[q].w - mean) * isd);
    // written once, never re-read by the GPU: write-rinse out of L2
    __builtin_nontemporal_store(y, (v4f*)op + q);
  }
}

// ---------------------------------------------------------------------------
// Launch
// ---------------------------------------------------------------------------
extern "C" void kernel_launch(void* const* d_in, const int* in_sizes, int n_in,
                              void* d_out, int out_size, void* d_ws,
                              size_t ws_size, hipStream_t stream) {
  const float* h   = (const float*)d_in[0];   // [8,2048,128] f32
  const int*   adj = (const int*)d_in[1];     // [8,2048,2048] i32
  const float* Ww  = (const float*)d_in[2];   // [128,128] f32
  const float* Wb  = (const float*)d_in[3];   // [128] f32
  const float* aw  = (const float*)d_in[4];   // [256] f32
  float* out = (float*)d_out;                 // [8,2048,128] f32

  // workspace layout (bytes)
  char* ws = (char*)d_ws;
  float*    u1  = (float*)(ws + 0);                    // 512 B
  float*    u2  = (float*)(ws + 512);                  // 512 B
  float*    c   = (float*)(ws + 1024);                 // 8 B
  _Float16* W16 = (_Float16*)(ws + 4096);              // 32 KB [128,128] f16
  float*    wh1 = (float*)(ws + 36864);                // 64 KB
  float*    wh2 = (float*)(ws + 102400);               // 64 KB
  _Float16* whT = (_Float16*)(ws + 167936);            // 4 MB [B,128,N] f16

  gat_prep_kernel<<<1, 128, 0, stream>>>(Ww, Wb, aw, u1, u2, c, W16);
  gat_wh_kernel<<<(B_ * N_ / 16) / 4, 128, 0, stream>>>(h, W16, Wb, whT);
  gat_score_kernel<<<(B_ * N_) / 256, 256, 0, stream>>>(h, u1, u2, c, wh1, wh2);
  gat_attn_kernel<<<(B_ * N_ / 16) / 4, 128, 0, stream>>>(adj, wh1, wh2, whT,
                                                          out);
}
